// Decoder_13357348290667
// MI455X (gfx1250) — matrix operand order repaired and checked by reading, not compiled, not run on hardware
//
#include <hip/hip_runtime.h>
#include <hip/hip_bf16.h>
#include <math.h>

// ---------- types for WMMA fragments (per CDNA5 ISA 7.12.2) ----------
typedef __bf16 bf16_t;
typedef __attribute__((ext_vector_type(16))) __bf16        v16bf;
typedef __attribute__((ext_vector_type(8)))  float         v8f;
typedef __attribute__((ext_vector_type(4)))  unsigned int  v4u;

union ABFrag { v16bf v; v4u q[2]; };   // 32 bytes = 16 bf16 = one A or B fragment per lane

#define B_   4
#define T_   1024
#define D_   512
#define V_   32000
#define L_   4
#define H_   8
#define HD_  64
#define M_   (B_ * T_)     // 4096 rows
#define BH_  (B_ * H_)     // 32 (batch*heads)

// ---------------------------------------------------------------------
// fp32 -> bf16 weight conversion (once per launch, deterministic)
// ---------------------------------------------------------------------
__global__ void k_cvt_bf16(const float* __restrict__ s, bf16_t* __restrict__ d, int n) {
    int i = blockIdx.x * blockDim.x + threadIdx.x;
    if (i < n) d[i] = (bf16_t)s[i];
}

// ---------------------------------------------------------------------
// x = W_emb[ids] + pos   (f32 master activations)
// ---------------------------------------------------------------------
__global__ void k_embed(const int* __restrict__ ids, const float* __restrict__ Wemb,
                        const float* __restrict__ pos, float* __restrict__ x) {
    size_t i  = (size_t)blockIdx.x * blockDim.x + threadIdx.x;   // over M_*D_
    int    d  = (int)(i & (D_ - 1));
    int    bt = (int)(i >> 9);
    int    t  = bt & (T_ - 1);
    int    tk = ids[bt];
    x[i] = Wemb[(size_t)tk * D_ + d] + pos[(size_t)t * D_ + d];
}

// ---------------------------------------------------------------------
// LayerNorm over D=512, f32 in -> bf16 out. 1 block (128 thr) per row.
// ---------------------------------------------------------------------
__global__ void k_layernorm(const float* __restrict__ x, const float* __restrict__ g,
                            const float* __restrict__ b, bf16_t* __restrict__ out) {
    const int row = blockIdx.x;
    const float* xr = x + (size_t)row * D_;
    float v[4];
    float s = 0.f;
#pragma unroll
    for (int i = 0; i < 4; ++i) { v[i] = xr[threadIdx.x + 128 * i]; s += v[i]; }
    __shared__ float red[128];
    red[threadIdx.x] = s; __syncthreads();
    for (int off = 64; off > 0; off >>= 1) {
        if ((int)threadIdx.x < off) red[threadIdx.x] += red[threadIdx.x + off];
        __syncthreads();
    }
    const float mu = red[0] * (1.0f / D_);
    __syncthreads();
    float vs = 0.f;
#pragma unroll
    for (int i = 0; i < 4; ++i) { float dd = v[i] - mu; vs += dd * dd; }
    red[threadIdx.x] = vs; __syncthreads();
    for (int off = 64; off > 0; off >>= 1) {
        if ((int)threadIdx.x < off) red[threadIdx.x] += red[threadIdx.x + off];
        __syncthreads();
    }
    const float rstd = rsqrtf(red[0] * (1.0f / D_) + 1e-5f);
#pragma unroll
    for (int i = 0; i < 4; ++i) {
        int c = threadIdx.x + 128 * i;
        out[(size_t)row * D_ + c] = (bf16_t)((v[i] - mu) * rstd * g[c] + b[c]);
    }
}

// ---------------------------------------------------------------------
// Generic WMMA GEMM: C[M,N] = A[M,K](bf16) * W[N,K]^T(bf16) + bias
// K and MODE are compile-time: every fragment address is a single base
// register + immediate offset (no per-iteration 64-bit VALU address math,
// no WMMA->VALU hazard NOPs). All 12 b128 loads of a K-step issue before
// the 8 WMMAs so the scheduler can clause them + use partial loadcnt waits.
// block = 256 thr = 8 waves (2 M-waves x 4 N-waves); wave tile 32x64;
// block tile 64(M) x 256(N). MODE: 0 = f32 (+resid), 1 = GELU->bf16, 2 = bf16
// ---------------------------------------------------------------------
template<int K, int MODE>
__global__ void k_gemm(const bf16_t* __restrict__ A, const bf16_t* __restrict__ W,
                       const float* __restrict__ bias, const float* __restrict__ resid,
                       float* __restrict__ outf, bf16_t* __restrict__ outb, int N) {
    const int lane = threadIdx.x & 31;
    const int w    = threadIdx.x >> 5;
    const int m0   = blockIdx.x * 64  + (w & 1) * 32;
    const int n0   = blockIdx.y * 256 + (w >> 1) * 64;
    const int row  = lane & 15;
    const int half = lane >> 4;

    const bf16_t* ab = A + (size_t)(m0 + row) * K + half * 8;   // A frags: +16*K for rows 16..31
    const bf16_t* wb = W + (size_t)(n0 + row) * K + half * 8;   // B frags: +nb*16*K (immediates)

    v8f acc[2][4] = {};
    for (int kk = 0; kk < K; kk += 32) {
        ABFrag a0, a1, bb[4];
        a0.q[0] = *(const v4u*)(ab + kk);
        a0.q[1] = *(const v4u*)(ab + kk + 16);
        a1.q[0] = *(const v4u*)(ab + 16 * K + kk);
        a1.q[1] = *(const v4u*)(ab + 16 * K + kk + 16);
#pragma unroll
        for (int nb = 0; nb < 4; ++nb) {
            bb[nb].q[0] = *(const v4u*)(wb + nb * 16 * K + kk);
            bb[nb].q[1] = *(const v4u*)(wb + nb * 16 * K + kk + 16);
        }
        __builtin_prefetch(ab + kk + 64, 0, 3);
        __builtin_prefetch(ab + 16 * K + kk + 64, 0, 3);
#pragma unroll
        for (int nb = 0; nb < 4; ++nb) {
            acc[0][nb] = __builtin_amdgcn_wmma_f32_16x16x32_bf16(
                false, a0.v, false, bb[nb].v, (short)0, acc[0][nb], false, false);
            acc[1][nb] = __builtin_amdgcn_wmma_f32_16x16x32_bf16(
                false, a1.v, false, bb[nb].v, (short)0, acc[1][nb], false, false);
        }
    }

#pragma unroll
    for (int nb = 0; nb < 4; ++nb) {
        const int n  = n0 + nb * 16 + (lane & 15);
        const float bv = bias ? bias[n] : 0.f;
#pragma unroll
        for (int mi = 0; mi < 2; ++mi) {
#pragma unroll
            for (int j = 0; j < 8; ++j) {
                const int m = m0 + mi * 16 + half * 8 + j;
                float vv = acc[mi][nb][j] + bv;
                const size_t off = (size_t)m * N + n;
                if (MODE == 0) {
                    if (resid) vv += resid[off];
                    outf[off] = vv;
                } else if (MODE == 1) {
                    outb[off] = (bf16_t)(0.5f * vv * (1.0f + erff(vv * 0.70710678118f)));
                } else {
                    outb[off] = (bf16_t)vv;
                }
            }
        }
    }
}

// ---------------------------------------------------------------------
// scores[bh][q][k] = scale * (Q . K), causal-masked to -1e30.
// Lower-triangle 16x16 tiles only; wave per tile; K-dim = hd = 64 (2 WMMA).
// grid (qt=64, ktgroup=8, bh=32), block 256 (8 waves)
// ---------------------------------------------------------------------
__global__ void k_scores(const bf16_t* __restrict__ qkv, float* __restrict__ scores) {
    const int lane = threadIdx.x & 31;
    const int w    = threadIdx.x >> 5;
    const int qt   = blockIdx.x;
    const int kt   = blockIdx.y * 8 + w;
    const int bh   = blockIdx.z;
    if (kt > qt) return;                       // wave-uniform: EXEC stays all-ones
    const int b = bh >> 3, h = bh & 7;
    const int row = lane & 15, half = lane >> 4;

    const bf16_t* qb = qkv + ((size_t)(b * T_ + qt * 16 + row)) * (3 * D_) + h * HD_ + half * 8;
    const bf16_t* kb = qkv + ((size_t)(b * T_ + kt * 16 + row)) * (3 * D_) + D_ + h * HD_ + half * 8;

    v8f acc = {};
#pragma unroll
    for (int kk = 0; kk < HD_; kk += 32) {
        ABFrag a, bb;
        a.q[0]  = *(const v4u*)(qb + kk);
        a.q[1]  = *(const v4u*)(qb + kk + 16);
        bb.q[0] = *(const v4u*)(kb + kk);
        bb.q[1] = *(const v4u*)(kb + kk + 16);
        acc = __builtin_amdgcn_wmma_f32_16x16x32_bf16(
            false, a.v, false, bb.v, (short)0, acc, false, false);
    }

    const float scale = 0.04419417382415922f;  // 1/sqrt(D=512), as in reference
#pragma unroll
    for (int j = 0; j < 8; ++j) {
        const int q = qt * 16 + half * 8 + j;
        const int k = kt * 16 + (lane & 15);
        float s = acc[j] * scale;
        if (k > q) s = -1e30f;
        scores[((size_t)bh * T_ + q) * T_ + k] = s;
    }
}

// ---------------------------------------------------------------------
// Causal softmax per row; reads k < kread = ((q>>4)+1)*16 (written region),
// writes bf16 P zero-padded to kfill = ((q>>5)+1)*32 so PV never reads poison.
// grid (q=1024, bh=32), block 128
// ---------------------------------------------------------------------
__global__ void k_softmax(const float* __restrict__ scores, bf16_t* __restrict__ P) {
    const int q  = blockIdx.x;
    const int bh = blockIdx.y;
    const int kread = ((q >> 4) + 1) * 16;
    const int kfill = ((q >> 5) + 1) * 32;
    const float* srow = scores + ((size_t)bh * T_ + q) * T_;
    bf16_t*      prow = P      + ((size_t)bh * T_ + q) * T_;

    float vals[8];
    float m = -3.0e38f;
    int   n = 0;
    for (int k = threadIdx.x; k < kread; k += 128) { float s = srow[k]; vals[n++] = s; m = fmaxf(m, s); }

    __shared__ float red[128];
    red[threadIdx.x] = m; __syncthreads();
    for (int off = 64; off > 0; off >>= 1) {
        if ((int)threadIdx.x < off) red[threadIdx.x] = fmaxf(red[threadIdx.x], red[threadIdx.x + off]);
        __syncthreads();
    }
    m = red[0]; __syncthreads();

    float s = 0.f;
    for (int i = 0; i < n; ++i) { vals[i] = __expf(vals[i] - m); s += vals[i]; }
    red[threadIdx.x] = s; __syncthreads();
    for (int off = 64; off > 0; off >>= 1) {
        if ((int)threadIdx.x < off) red[threadIdx.x] += red[threadIdx.x + off];
        __syncthreads();
    }
    const float inv = 1.0f / red[0];

    n = 0;
    for (int k = threadIdx.x; k < kfill; k += 128)
        prow[k] = (k < kread) ? (bf16_t)(vals[n++] * inv) : (bf16_t)0.0f;
}

// ---------------------------------------------------------------------
// Vt[bh][d][t] = V[b][t][h][d]  (so PV's B-fragment is K(t)-contiguous)
// ---------------------------------------------------------------------
__global__ void k_vtrans(const bf16_t* __restrict__ qkv, bf16_t* __restrict__ Vt) {
    size_t i = (size_t)blockIdx.x * blockDim.x + threadIdx.x;   // BH*HD*T
    const int t  = (int)(i & (T_ - 1));
    const int d  = (int)((i >> 10) & (HD_ - 1));
    const int bh = (int)(i >> 16);
    const int b = bh >> 3, h = bh & 7;
    Vt[i] = qkv[((size_t)(b * T_ + t)) * (3 * D_) + 2 * D_ + h * HD_ + d];
}

// ---------------------------------------------------------------------
// Z[b][q][h*64+d] = P @ V.  Wave owns a 32-query strip (2 A-frags reused
// over 4 B-frags; all strides compile-time -> immediate offsets).
// Causal K extent is uniform across the strip: nk = qp+1 (wave-uniform).
// grid (qpgroup=4, 1, bh=32), block 256 (8 waves)
// ---------------------------------------------------------------------
__global__ void k_attn_pv(const bf16_t* __restrict__ P, const bf16_t* __restrict__ Vt,
                          bf16_t* __restrict__ Z) {
    const int lane = threadIdx.x & 31;
    const int w    = threadIdx.x >> 5;
    const int qp   = blockIdx.x * 8 + w;       // 0..31 (32-row strip index)
    const int bh   = blockIdx.z;
    const int b = bh >> 3, h = bh & 7;
    const int row = lane & 15, half = lane >> 4;

    const bf16_t* prow  = P  + ((size_t)bh * T_ + qp * 32 + row) * T_ + half * 8;
    const bf16_t* vbase = Vt + ((size_t)bh * HD_ + row) * T_ + half * 8;

    v8f acc[2][4] = {};
    const int nk = qp + 1;                     // number of 32-wide K chunks (causal)
    for (int c = 0; c < nk; ++c) {
        const int k0 = c * 32;
        ABFrag a0, a1, bb[4];
        a0.q[0] = *(const v4u*)(prow + k0);
        a0.q[1] = *(const v4u*)(prow + k0 + 16);
        a1.q[0] = *(const v4u*)(prow + 16 * T_ + k0);
        a1.q[1] = *(const v4u*)(prow + 16 * T_ + k0 + 16);
#pragma unroll
        for (int nb = 0; nb < 4; ++nb) {
            bb[nb].q[0] = *(const v4u*)(vbase + nb * 16 * T_ + k0);
            bb[nb].q[1] = *(const v4u*)(vbase + nb * 16 * T_ + k0 + 16);
        }
#pragma unroll
        for (int nb = 0; nb < 4; ++nb) {
            acc[0][nb] = __builtin_amdgcn_wmma_f32_16x16x32_bf16(
                false, a0.v, false, bb[nb].v, (short)0, acc[0][nb], false, false);
            acc[1][nb] = __builtin_amdgcn_wmma_f32_16x16x32_bf16(
                false, a1.v, false, bb[nb].v, (short)0, acc[1][nb], false, false);
        }
    }

#pragma unroll
    for (int nb = 0; nb < 4; ++nb) {
        const int d = nb * 16 + (lane & 15);
#pragma unroll
        for (int mi = 0; mi < 2; ++mi) {
#pragma unroll
            for (int j = 0; j < 8; ++j) {
                const int q = qp * 32 + mi * 16 + half * 8 + j;
                Z[((size_t)(b * T_ + q)) * D_ + h * HD_ + d] = (bf16_t)acc[mi][nb][j];
            }
        }
    }
}

// ---------------------------------------------------------------------
// host-side orchestration
// ---------------------------------------------------------------------
extern "C" void kernel_launch(void* const* d_in, const int* in_sizes, int n_in,
                              void* d_out, int out_size, void* d_ws, size_t ws_size,
                              hipStream_t stream) {
    (void)in_sizes; (void)n_in; (void)out_size; (void)ws_size;

    const int*   ids   = (const int*)  d_in[0];
    const float* Wemb  = (const float*)d_in[1];
    const float* pos   = (const float*)d_in[2];
    const float* Wqkv  = (const float*)d_in[3];
    const float* bqkv  = (const float*)d_in[4];
    const float* Wo    = (const float*)d_in[5];
    const float* bo    = (const float*)d_in[6];
    const float* ln1g  = (const float*)d_in[7];
    const float* ln1b  = (const float*)d_in[8];
    const float* ln2g  = (const float*)d_in[9];
    const float* ln2b  = (const float*)d_in[10];
    const float* Wh    = (const float*)d_in[11];
    const float* bh    = (const float*)d_in[12];
    const float* Wmo   = (const float*)d_in[13];
    const float* bmo   = (const float*)d_in[14];
    const float* lnfg  = (const float*)d_in[15];
    const float* lnfb  = (const float*)d_in[16];
    float* logits = (float*)d_out;

    // workspace layout
    char* p = (char*)d_ws;
    auto alloc = [&](size_t bytes) { char* r = p; p += (bytes + 255) & ~(size_t)255; return r; };

    const int cWq  = L_ * 3 * D_ * D_;       // 3,145,728
    const int cWo  = L_ * D_ * D_;           // 1,048,576
    const int cWh  = L_ * 4 * D_ * D_;       // 4,194,304
    const int cWmo = L_ * D_ * 4 * D_;       // 4,194,304
    const int cWe  = V_ * D_;                // 16,384,000

    bf16_t* wqB  = (bf16_t*)alloc((size_t)cWq  * 2);
    bf16_t* woB  = (bf16_t*)alloc((size_t)cWo  * 2);
    bf16_t* whB  = (bf16_t*)alloc((size_t)cWh  * 2);
    bf16_t* wmoB = (bf16_t*)alloc((size_t)cWmo * 2);
    bf16_t* weB  = (bf16_t*)alloc((size_t)cWe  * 2);
    float*  xbuf = (float*) alloc((size_t)M_ * D_ * 4);
    float*  zbuf = (float*) alloc((size_t)M_ * D_ * 4);
    bf16_t* lno  = (bf16_t*)alloc((size_t)M_ * D_ * 2);
    bf16_t* qkvB = (bf16_t*)alloc((size_t)M_ * 3 * D_ * 2);
    float*  scoB = (float*) alloc((size_t)BH_ * T_ * T_ * 4);
    bf16_t* Pb   = (bf16_t*)alloc((size_t)BH_ * T_ * T_ * 2);
    bf16_t* VtB  = (bf16_t*)alloc((size_t)BH_ * HD_ * T_ * 2);
    bf16_t* zatt = (bf16_t*)alloc((size_t)M_ * D_ * 2);
    bf16_t* hbuf = (bf16_t*)alloc((size_t)M_ * 4 * D_ * 2);

    // --- weight conversion (every launch: deterministic, no caching) ---
    k_cvt_bf16<<<(cWq  + 255) / 256, 256, 0, stream>>>(Wqkv, wqB,  cWq);
    k_cvt_bf16<<<(cWo  + 255) / 256, 256, 0, stream>>>(Wo,   woB,  cWo);
    k_cvt_bf16<<<(cWh  + 255) / 256, 256, 0, stream>>>(Wh,   whB,  cWh);
    k_cvt_bf16<<<(cWmo + 255) / 256, 256, 0, stream>>>(Wmo,  wmoB, cWmo);
    k_cvt_bf16<<<(cWe  + 255) / 256, 256, 0, stream>>>(Wemb, weB,  cWe);

    // --- embedding ---
    k_embed<<<(M_ * D_) / 256, 256, 0, stream>>>(ids, Wemb, pos, xbuf);

    // --- transformer layers ---
    for (int l = 0; l < L_; ++l) {
        const bf16_t* wq_l  = wqB  + (size_t)l * 3 * D_ * D_;
        const bf16_t* wo_l  = woB  + (size_t)l * D_ * D_;
        const bf16_t* wh_l  = whB  + (size_t)l * 4 * D_ * D_;
        const bf16_t* wmo_l = wmoB + (size_t)l * D_ * 4 * D_;

        // ln1(x) -> bf16
        k_layernorm<<<M_, 128, 0, stream>>>(xbuf, ln1g + l * D_, ln1b + l * D_, lno);
        // qkv = ln1 @ Wqkv^T + b  (bf16 out)
        k_gemm<D_, 2><<<dim3(M_ / 64, (3 * D_) / 256), 256, 0, stream>>>(
            lno, wq_l, bqkv + (size_t)l * 3 * D_, nullptr, nullptr, qkvB, 3 * D_);
        // attention core
        k_scores <<<dim3(T_ / 16, 8, BH_), 256, 0, stream>>>(qkvB, scoB);
        k_softmax<<<dim3(T_, BH_),          128, 0, stream>>>(scoB, Pb);
        k_vtrans <<<(BH_ * HD_ * T_) / 256, 256, 0, stream>>>(qkvB, VtB);
        k_attn_pv<<<dim3(4, 1, BH_),        256, 0, stream>>>(Pb, VtB, zatt);
        // z = x + attn @ Wo^T + bo   (f32)
        k_gemm<D_, 0><<<dim3(M_ / 64, D_ / 256), 256, 0, stream>>>(
            zatt, wo_l, bo + (size_t)l * D_, xbuf, zbuf, nullptr, D_);
        // ln2(z) -> bf16
        k_layernorm<<<M_, 128, 0, stream>>>(zbuf, ln2g + l * D_, ln2b + l * D_, lno);
        // h = gelu(ln2 @ Wh^T + bh)  (bf16)
        k_gemm<D_, 1><<<dim3(M_ / 64, (4 * D_) / 256), 256, 0, stream>>>(
            lno, wh_l, bh + (size_t)l * 4 * D_, nullptr, nullptr, hbuf, 4 * D_);
        // x = z + h @ Wmo^T + bmo    (f32)
        k_gemm<4 * D_, 0><<<dim3(M_ / 64, D_ / 256), 256, 0, stream>>>(
            hbuf, wmo_l, bmo + (size_t)l * D_, zbuf, xbuf, nullptr, D_);
    }

    // --- final LN + tied lm_head ---
    k_layernorm<<<M_, 128, 0, stream>>>(xbuf, lnfg, lnfb, lno);
    k_gemm<D_, 0><<<dim3(M_ / 64, V_ / 256), 256, 0, stream>>>(
        lno, weB, nullptr, nullptr, logits, nullptr, V_);
}
